// Decoder_5729486372912
// MI455X (gfx1250) — matrix operand order repaired and checked by reading, not compiled, not run on hardware
//
#include <hip/hip_runtime.h>
#include <hip/hip_bf16.h>

typedef __attribute__((ext_vector_type(16))) __bf16 v16bf;
typedef __attribute__((ext_vector_type(8)))  float  v8f;
typedef __attribute__((ext_vector_type(4)))  unsigned int u32x4;
typedef __attribute__((ext_vector_type(8)))  int i32x8;
typedef __attribute__((ext_vector_type(4)))  int i32x4;

#define HDIM     256
#define NGATES   1024      // 4*H
#define TSTEPS   80
#define WG_ROWS  128
#define NROWS    8192      // B*A

#if defined(__has_builtin)
#if __has_builtin(__builtin_amdgcn_tensor_load_to_lds) && __has_builtin(__builtin_amdgcn_s_wait_tensorcnt)
#define HAVE_TDM 1
#endif
#endif
#ifndef HAVE_TDM
#define HAVE_TDM 0
#endif

__device__ __forceinline__ float fsig(float x) {
    return __builtin_amdgcn_rcpf(1.0f + __expf(-x));
}
__device__ __forceinline__ float ftanh(float x) {
    // tanh(x) = 1 - 2/(exp(2x)+1)
    return 1.0f - 2.0f * __builtin_amdgcn_rcpf(1.0f + __expf(2.0f * x));
}

#if HAVE_TDM
// 1-D TDM copy: 32KB (8192 dwords) global -> LDS.  D# per CDNA5 ISA ch.8:
// g0: count=1 | lds_addr | global_addr[56:0] | type=2
// g1: data_size=4B, tensor_dim0=tile_dim0=8192 dwords, tensor_dim1=tile_dim1=1
__device__ __forceinline__ void tdm_load_chunk(const void* gsrc, unsigned lds_addr) {
    unsigned long long ga = (unsigned long long)(uintptr_t)gsrc;
    union { unsigned u[4]; u32x4 v; } g0;
    g0.u[0] = 1u;                                               // count=1
    g0.u[1] = lds_addr;                                         // lds byte addr
    g0.u[2] = (unsigned)(ga & 0xFFFFFFFFu);                     // global[31:0]
    g0.u[3] = (unsigned)((ga >> 32) & 0x1FFFFFFu) | (2u << 30); // global[56:32] | type=2
    union { int i[8]; i32x8 v; } g1;
    g1.i[0] = (int)(2u << 16);          // workgroup_mask=0, data_size=2 (4B)
    g1.i[1] = (int)(8192u << 16);       // tensor_dim0[15:0]=8192  (dwords)
    g1.i[2] = (int)(1u << 16);          // tensor_dim0[31:16]=0, tensor_dim1=1
    g1.i[3] = (int)(8192u << 16);       // tensor_dim1[31:16]=0, tile_dim0=8192
    g1.i[4] = 1;                        // tile_dim1=1, tile_dim2=0
    g1.i[5] = 8192;                     // tensor_dim0_stride
    g1.i[6] = 0;
    g1.i[7] = 0;
    i32x4 z4 = (i32x4){0, 0, 0, 0};
#if __clang_major__ >= 23
    i32x8 z8 = (i32x8){0, 0, 0, 0, 0, 0, 0, 0};
    __builtin_amdgcn_tensor_load_to_lds(g0.v, g1.v, z4, z4, z8, 0);
#else
    __builtin_amdgcn_tensor_load_to_lds(g0.v, g1.v, z4, z4, 0);
#endif
}
#endif

// ---------------------------------------------------------------------------
// Pre-pack W_hh (f32 [1024][256]) into bf16 WMMA B-fragment order:
//   idx = (((t*4 + gate)*8 + ks)*32 + lane)*16 + e
//   value = W_hh[gate*256 + t*16 + (lane&15)][ks*32 + (lane>>4)*16 + e]
// One contiguous 32KB chunk per t (16 chunks = 512KB in d_ws).
// ---------------------------------------------------------------------------
__global__ void prepack_whh(const float* __restrict__ Whh, __bf16* __restrict__ wp) {
    int i    = blockIdx.x * 256 + threadIdx.x;      // 262144 elements
    int e    = i & 15;
    int L    = (i >> 4) & 31;
    int ks   = (i >> 9) & 7;
    int gate = (i >> 12) & 3;
    int t    = (i >> 14) & 15;
    int n = gate * 256 + t * 16 + (L & 15);         // gate-output column
    int k = ks * 32 + (L >> 4) * 16 + e;            // reduction index
    wp[i] = (__bf16)Whh[n * HDIM + k];
}

// ---------------------------------------------------------------------------
// Persistent LSTM decoder: 64 workgroups x 256 threads (8 waves).
// Wave w owns rows [wg*128 + w*16, +16). h (bf16), c (f32) live in LDS.
// ---------------------------------------------------------------------------
__launch_bounds__(256, 1)
__global__ void lstm_decoder(const float* __restrict__ pos,
                             const unsigned char* __restrict__ avail,
                             const float* __restrict__ hidden,
                             const float* __restrict__ context,
                             const float* __restrict__ Wih,
                             const float* __restrict__ bih,
                             const float* __restrict__ bhh,
                             const float* __restrict__ Wlin,
                             const float* __restrict__ blin,
                             const __bf16* __restrict__ wpack,
                             float* __restrict__ out) {
    extern __shared__ char smem[];
    __bf16* hL   = (__bf16*)(smem);                    //  65536 B : [128][256] bf16
    float*  cL   = (float*) (smem + 65536);            // 131072 B : [128][256] f32
    __bf16* bbuf = (__bf16*)(smem + 196608);           //  65536 B : 2 x 32KB B chunks
    float*  bsum = (float*) (smem + 262144);           //   4096 B : b_ih + b_hh [1024]
    float*  wih  = (float*) (smem + 266240);           //   8192 B : W_ih [1024][2]
    float*  wlin = (float*) (smem + 274432);           //   2048 B : W_lin [2][256]
    float*  outx = (float*) (smem + 276480);           //   1024 B : output [128][2]
    float*  blinL= (float*) (smem + 277504);           //      8 B

    const int tid     = threadIdx.x;
    const int rowbase = blockIdx.x * WG_ROWS;

    // ---- init LDS state -------------------------------------------------
    for (int i = tid; i < WG_ROWS * HDIM; i += 256) {
        hL[i] = (__bf16)hidden[(size_t)rowbase * HDIM + i];
        cL[i] = context[(size_t)rowbase * HDIM + i];
    }
    for (int i = tid; i < NGATES; i += 256)      bsum[i] = bih[i] + bhh[i];
    for (int i = tid; i < NGATES * 2; i += 256)  wih[i]  = Wih[i];
    for (int i = tid; i < 2 * HDIM; i += 256)    wlin[i] = Wlin[i];
    for (int i = tid; i < WG_ROWS * 2; i += 256) outx[i] = pos[(size_t)rowbase * 2 + i];
    if (tid < 2) blinL[tid] = blin[tid];

    const int lane   = tid & 31;
    const int wave   = tid >> 5;
    const int laneHi = lane >> 4;       // 0/1 : C-layout M offset 0/8
    const int nIdx   = lane & 15;       // C-layout N (column) lane
    const int mloc   = wave * 16;       // wave's row-tile base within WG

    float maskr[8];
    #pragma unroll
    for (int r = 0; r < 8; ++r)
        maskr[r] = avail[rowbase + mloc + r + 8 * laneHi] ? 1.0f : 0.0f;

    const unsigned bbuf_lds = (unsigned)(uintptr_t)(void*)bbuf;  // LDS byte addr
#if !HAVE_TDM
    const uint4* wp4 = (const uint4*)wpack;     // chunk t = 2048 uint4 @ t*2048
    uint4 stg[8];
#endif

    // ---- prologue: start DMA of chunk 0 into bbuf[0] --------------------
#if HAVE_TDM
    if (wave == 0)
        tdm_load_chunk((const char*)wpack, bbuf_lds);
    __syncthreads();                            // init-LDS visible to all
#else
    #pragma unroll
    for (int j = 0; j < 8; ++j) stg[j] = wp4[j * 256 + tid];
    __syncthreads();
    {
        uint4* dst = (uint4*)bbuf;
        #pragma unroll
        for (int j = 0; j < 8; ++j) dst[j * 256 + tid] = stg[j];
    }
#endif

    // ---- time recurrence ------------------------------------------------
    for (int ts = 0; ts < TSTEPS; ++ts) {
        // Hoist A fragments (own 16 h-rows) for all 8 k-steps.
        // A order used here (contiguous k): lane M = lane%16; VGPR0-7 = K[base..base+16),
        // base = ks*32 + laneHi*16 (matches packed B order).
        v16bf afr[8];
        #pragma unroll
        for (int ks = 0; ks < 8; ++ks) {
            int kb = ks * 32 + laneHi * 16;
            union { uint4 u[2]; v16bf v; } au;
            au.u[0] = *(const uint4*)(hL + (mloc + nIdx) * HDIM + kb);
            au.u[1] = *(const uint4*)(hL + (mloc + nIdx) * HDIM + kb + 8);
            afr[ks] = au.v;
        }
        // current output (x) for this lane's 8 rows
        float x0[8], x1[8], dl0[8], dl1[8];
        #pragma unroll
        for (int r = 0; r < 8; ++r) {
            int m = mloc + r + 8 * laneHi;
            x0[r] = outx[m * 2 + 0];
            x1[r] = outx[m * 2 + 1];
            dl0[r] = 0.0f; dl1[r] = 0.0f;
        }

        for (int t = 0; t < 16; ++t) {          // 16 column-groups of 64 gate cols
            int nc = (t + 1) & 15;              // next chunk (wraps for next ts)
#if HAVE_TDM
            __builtin_amdgcn_s_wait_tensorcnt(0);   // chunk t landed (wave0; NOP others)
            __syncthreads();                        // everyone past reads of buf[(t+1)&1]
            if (wave == 0)                          // DMA next chunk, overlaps compute
                tdm_load_chunk((const char*)wpack + (size_t)nc * 32768,
                               bbuf_lds + (unsigned)(((t + 1) & 1) * 32768));
#else
            #pragma unroll
            for (int j = 0; j < 8; ++j) stg[j] = wp4[nc * 2048 + j * 256 + tid];
            __syncthreads();
#endif
            const __bf16* bch = bbuf + (t & 1) * 16384;

            v8f acc[4];
            #pragma unroll
            for (int g = 0; g < 4; ++g)
                acc[g] = (v8f){0.f,0.f,0.f,0.f,0.f,0.f,0.f,0.f};

            #pragma unroll
            for (int ks = 0; ks < 8; ++ks) {
                #pragma unroll
                for (int g = 0; g < 4; ++g) {
                    union { uint4 u[2]; v16bf v; } bu;
                    const uint4* bp = (const uint4*)(bch + ((g * 8 + ks) * 512 + lane * 16));
                    bu.u[0] = bp[0];
                    bu.u[1] = bp[1];
                    acc[g] = __builtin_amdgcn_wmma_f32_16x16x32_bf16(
                        false, afr[ks], false, bu.v, (short)0, acc[g], false, false);
                }
            }

            // ---- pointwise LSTM update for col = t*16 + nIdx ------------
            int col = t * 16 + nIdx;
            float bsi = bsum[col],        bsf = bsum[256 + col];
            float bsg = bsum[512 + col],  bso = bsum[768 + col];
            float wi0i = wih[col * 2],           wi1i = wih[col * 2 + 1];
            float wi0f = wih[(256 + col) * 2],   wi1f = wih[(256 + col) * 2 + 1];
            float wi0g = wih[(512 + col) * 2],   wi1g = wih[(512 + col) * 2 + 1];
            float wi0o = wih[(768 + col) * 2],   wi1o = wih[(768 + col) * 2 + 1];
            float wl0 = wlin[col], wl1 = wlin[HDIM + col];

            #pragma unroll
            for (int r = 0; r < 8; ++r) {
                int m = mloc + r + 8 * laneHi;
                float ri = acc[0][r] + bsi + x0[r] * wi0i + x1[r] * wi1i;
                float rf = acc[1][r] + bsf + x0[r] * wi0f + x1[r] * wi1f;
                float rg = acc[2][r] + bsg + x0[r] * wi0g + x1[r] * wi1g;
                float ro = acc[3][r] + bso + x0[r] * wi0o + x1[r] * wi1o;
                float iv = fsig(ri), fv = fsig(rf), ov = fsig(ro);
                float gv = ftanh(rg);
                float cold = cL[m * HDIM + col];
                float cn = fv * cold + iv * gv;
                float hn = ov * ftanh(cn);
                float hold = (float)hL[m * HDIM + col];
                float hv = (maskr[r] > 0.0f) ? hn : hold;
                if (maskr[r] > 0.0f) {
                    cL[m * HDIM + col] = cn;
                    hL[m * HDIM + col] = (__bf16)hn;
                }
                dl0[r] += hv * wl0;             // output-projection partials
                dl1[r] += hv * wl1;
            }

#if !HAVE_TDM
            {   // store prefetched chunk for t+1 into the other buffer
                uint4* dst = (uint4*)(bbuf + ((t + 1) & 1) * 16384);
                #pragma unroll
                for (int j = 0; j < 8; ++j) dst[j * 256 + tid] = stg[j];
            }
#endif
        }

        // ---- reduce delta over the 16 lanes sharing each row ------------
        #pragma unroll
        for (int r = 0; r < 8; ++r) {
            #pragma unroll
            for (int s = 1; s < 16; s <<= 1) {
                dl0[r] += __shfl_xor(dl0[r], s, 32);
                dl1[r] += __shfl_xor(dl1[r], s, 32);
            }
        }
        if (nIdx == 0) {                        // lanes 0 & 16 own row sums
            float bl0 = blinL[0], bl1 = blinL[1];
            #pragma unroll
            for (int r = 0; r < 8; ++r) {
                int m = mloc + r + 8 * laneHi;
                float o0 = outx[m * 2 + 0] + dl0[r] + bl0;
                float o1 = outx[m * 2 + 1] + dl1[r] + bl1;
                outx[m * 2 + 0] = o0;
                outx[m * 2 + 1] = o1;
                float* op = out + ((size_t)(rowbase + m) * TSTEPS + ts) * 2;
                op[0] = o0;
                op[1] = o1;
            }
        }
        // outx written/read only by the owning wave -> in-order DS, no barrier
    }
}

extern "C" void kernel_launch(void* const* d_in, const int* in_sizes, int n_in,
                              void* d_out, int out_size, void* d_ws, size_t ws_size,
                              hipStream_t stream) {
    (void)in_sizes; (void)n_in; (void)out_size; (void)ws_size;
    const float*         pos     = (const float*)d_in[0];
    const unsigned char* avail   = (const unsigned char*)d_in[1];
    const float*         hidden  = (const float*)d_in[2];
    const float*         context = (const float*)d_in[3];
    const float*         Wih     = (const float*)d_in[4];
    const float*         Whh     = (const float*)d_in[5];
    const float*         bih     = (const float*)d_in[6];
    const float*         bhh     = (const float*)d_in[7];
    const float*         Wlin    = (const float*)d_in[8];
    const float*         blin    = (const float*)d_in[9];
    __bf16* wpack = (__bf16*)d_ws;              // 512 KB packed W_hh

    prepack_whh<<<1024, 256, 0, stream>>>(Whh, wpack);

    size_t shmem = 278528;                      // ~272 KB of the 320 KB WGP LDS
    hipFuncSetAttribute((const void*)lstm_decoder,
                        hipFuncAttributeMaxDynamicSharedMemorySize, (int)shmem);
    lstm_decoder<<<NROWS / WG_ROWS, 256, shmem, stream>>>(
        pos, avail, hidden, context, Wih, bih, bhh, Wlin, blin, wpack,
        (float*)d_out);
}
